// MHSAModule_8847632629856
// MI455X (gfx1250) — compile-verified, matrix operand order checked by reading, not run on hardware
//
#include <hip/hip_runtime.h>
#include <hip/hip_bf16.h>

// MHSA with RoPE for MI455X (gfx1250), bf16 WMMA pipeline.
// B=8, S=1024, D=512, H=8, Dh=64, M=B*S=8192.

typedef __bf16 bf16_t;
typedef __attribute__((ext_vector_type(16))) __bf16 v16bf;
typedef __attribute__((ext_vector_type(8)))  float  v8f;

#define WMMA_BF16(a, b, c) \
  __builtin_amdgcn_wmma_f32_16x16x32_bf16(false, (a), false, (b), (short)0, (c), false, false)

// Forbid the scheduler from mixing fragment loads into the WMMA group.
#define SCHED_FENCE() __builtin_amdgcn_sched_barrier(0)

// Load a 16x32 bf16 fragment (A-layout; B uses the same per-lane layout when the
// tile is stored [n][k] row-major in LDS).  Lane L holds row (L&15); lanes 0-15
// hold K in {0..7, 16..23}, lanes 16-31 hold K in {8..15, 24..31}.
static __device__ inline v16bf frag_ld(const bf16_t* __restrict__ base, int ld) {
  int lane = threadIdx.x & 31;
  int r  = lane & 15;
  int kb = (lane >> 4) << 3;  // 0 or 8
  const bf16_t* p = base + r * ld + kb;
  v16bf f;
#pragma unroll
  for (int i = 0; i < 8; ++i) f[i] = p[i];
#pragma unroll
  for (int i = 0; i < 8; ++i) f[8 + i] = p[16 + i];
  return f;
}

// ---------------------------------------------------------------------------
// Kernel 1: fused QKV projection.  grid = (M/128, N/64, 3), block = 256 (8 waves).
// K-chunk = 64; software-pipelined global->LDS staging.  Out = bf16 [8192][512].
// ---------------------------------------------------------------------------
__global__ __launch_bounds__(256)
void gemm_qkv_kernel(const float* __restrict__ Xq, const float* __restrict__ Xk,
                     const float* __restrict__ Xv,
                     const float* __restrict__ Wq, const float* __restrict__ Wk,
                     const float* __restrict__ Wv,
                     bf16_t* __restrict__ Oq, bf16_t* __restrict__ Ok,
                     bf16_t* __restrict__ Ov) {
  const int N = 512, K = 512;
  const float* X;
  const float* W;
  bf16_t* O;
  if (blockIdx.z == 0)      { X = Xq; W = Wq; O = Oq; }
  else if (blockIdx.z == 1) { X = Xk; W = Wk; O = Ok; }
  else                      { X = Xv; W = Wv; O = Ov; }

  __shared__ bf16_t As[128 * 64];  // [m][k]  16 KB
  __shared__ bf16_t Bs[64 * 64];   // [n][k]   8 KB (transposed weight tile)

  const int bm = blockIdx.x * 128, bn = blockIdx.y * 64;
  const int tid = threadIdx.x, w = tid >> 5;

  const int arow = tid >> 1, akh = (tid & 1) * 32;   // A staging coords
  const int bnn = tid >> 2,  bkq = (tid & 3) * 16;   // B staging coords

  float4 fv[8];     // staged A chunk (32 fp32)
  float  wvf[16];   // staged B chunk (16 fp32, strided)

  // prologue: stage chunk 0 into registers
  {
    const float* g = X + (long)(bm + arow) * K + akh;
#pragma unroll
    for (int i = 0; i < 8; ++i) fv[i] = ((const float4*)g)[i];
#pragma unroll
    for (int i = 0; i < 16; ++i) wvf[i] = W[(long)(bkq + i) * N + bn + bnn];
  }

  v8f acc[4] = {};
  for (int k0 = 0; k0 < K; k0 += 64) {
    // ---- spill staged registers to LDS
    {
      bf16_t* s = As + arow * 64 + akh;
#pragma unroll
      for (int i = 0; i < 8; ++i) {
        s[4 * i + 0] = (bf16_t)fv[i].x;
        s[4 * i + 1] = (bf16_t)fv[i].y;
        s[4 * i + 2] = (bf16_t)fv[i].z;
        s[4 * i + 3] = (bf16_t)fv[i].w;
      }
      bf16_t* sb = Bs + bnn * 64 + bkq;
#pragma unroll
      for (int i = 0; i < 16; ++i) sb[i] = (bf16_t)wvf[i];
    }
    __syncthreads();

    // ---- issue global loads for chunk k0+64 (overlaps with WMMA below)
    if (k0 + 64 < K) {
      const float* g = X + (long)(bm + arow) * K + k0 + 64 + akh;
      __builtin_prefetch(g + 64, 0, 1);
#pragma unroll
      for (int i = 0; i < 8; ++i) fv[i] = ((const float4*)g)[i];
#pragma unroll
      for (int i = 0; i < 16; ++i)
        wvf[i] = W[(long)(k0 + 64 + bkq + i) * N + bn + bnn];
    }

    // ---- load ALL fragments, fence, then 8 WMMAs back-to-back
    v16bf af0 = frag_ld(As + (w * 16) * 64, 64);
    v16bf af1 = frag_ld(As + (w * 16) * 64 + 32, 64);
    v16bf bfr[8];
#pragma unroll
    for (int t = 0; t < 4; ++t) {
      bfr[2 * t]     = frag_ld(Bs + (t * 16) * 64, 64);
      bfr[2 * t + 1] = frag_ld(Bs + (t * 16) * 64 + 32, 64);
    }
    SCHED_FENCE();
#pragma unroll
    for (int t = 0; t < 4; ++t) acc[t] = WMMA_BF16(af0, bfr[2 * t], acc[t]);
#pragma unroll
    for (int t = 0; t < 4; ++t) acc[t] = WMMA_BF16(af1, bfr[2 * t + 1], acc[t]);
    SCHED_FENCE();
    __syncthreads();
  }

  const int lane = tid & 31, n = lane & 15, mo = (lane >> 4) << 3;
#pragma unroll
  for (int t = 0; t < 4; ++t)
#pragma unroll
    for (int r = 0; r < 8; ++r)
      O[(long)(bm + w * 16 + mo + r) * N + bn + t * 16 + n] = (bf16_t)acc[t][r];
}

// ---------------------------------------------------------------------------
// Kernel 2: RoPE on Q and K (in-place on bf16 workspace), Q also scaled 1/8.
// ---------------------------------------------------------------------------
__global__ void rope_kernel(bf16_t* __restrict__ Q, bf16_t* __restrict__ Kt) {
  const int total = 8 * 1024 * 8 * 32;  // B*S*H*half
  int idx = blockIdx.x * blockDim.x + threadIdx.x;
  if (idx >= total) return;
  int d   = idx & 31;
  int h   = (idx >> 5) & 7;
  int bsi = idx >> 8;              // b*S + s
  int s   = bsi & 1023;
  // inv_freq = 10000^(-d/32) = exp(-d * ln(10000)/32)
  float inv = __expf(-(float)d * (9.210340371976184f / 32.0f));
  float ang = (float)s * inv;
  float sn, cs;
  __sincosf(ang, &sn, &cs);
  long base = (long)bsi * 512 + h * 64 + d;
  {
    float x1 = (float)Q[base], x2 = (float)Q[base + 32];
    Q[base]      = (bf16_t)((x1 * cs - x2 * sn) * 0.125f);  // 1/sqrt(64)
    Q[base + 32] = (bf16_t)((x2 * cs + x1 * sn) * 0.125f);
  }
  {
    float x1 = (float)Kt[base], x2 = (float)Kt[base + 32];
    Kt[base]      = (bf16_t)(x1 * cs - x2 * sn);
    Kt[base + 32] = (bf16_t)(x2 * cs + x1 * sn);
  }
}

// ---------------------------------------------------------------------------
// Kernel 3: causal flash attention.  grid = (S/128, B*H), block = 256 (8 waves).
// Each wave owns a 16-row q tile (128 q rows / block); keys streamed in chunks
// of 32 through LDS shared by all 8 waves.
// ---------------------------------------------------------------------------
__global__ __launch_bounds__(256)
void flash_attn_kernel(const bf16_t* __restrict__ Q, const bf16_t* __restrict__ Kt,
                       const bf16_t* __restrict__ V, bf16_t* __restrict__ Oa) {
  const int S = 1024, HD = 512, Dh = 64;
  const int bh = blockIdx.y;
  const int b = bh >> 3, h = bh & 7;
  const int w = threadIdx.x >> 5, lane = threadIdx.x & 31;
  const int qblk = blockIdx.x * 128;
  const int q0 = qblk + w * 16;

  __shared__ bf16_t Ks[32 * 64];      // [key][d]            4 KB
  __shared__ bf16_t Vs[64 * 32];      // [d][key] transposed 4 KB
  __shared__ bf16_t Ps[8][16 * 32];   // per-wave P tiles    8 KB

  // Q fragments for the two 32-wide K-dim steps (head dim = 64), A-layout.
  const bf16_t* qbase = Q + ((long)(b * S) + q0) * HD + h * Dh;
  const int m = lane & 15, kb = (lane >> 4) << 3;
  v16bf qf0, qf1;
#pragma unroll
  for (int i = 0; i < 8; ++i) {
    qf0[i]     = qbase[m * HD + kb + i];
    qf0[8 + i] = qbase[m * HD + kb + 16 + i];
    qf1[i]     = qbase[m * HD + 32 + kb + i];
    qf1[8 + i] = qbase[m * HD + 32 + kb + 16 + i];
  }

  v8f o[4] = {};
  float mrow[8], lrow[8];
#pragma unroll
  for (int r = 0; r < 8; ++r) { mrow[r] = -1e30f; lrow[r] = 0.0f; }

  const int nchunks = (qblk + 128) >> 5;  // causal extent of the whole block
  for (int j = 0; j < nchunks; ++j) {
    const int kkey = j << 5;
    // ---- stage K chunk (32x64) and V chunk transposed (64x32); 256 threads
    {
      int key = threadIdx.x >> 3, dq = (threadIdx.x & 7) * 8;
      const bf16_t* gk = Kt + ((long)(b * S) + kkey + key) * HD + h * Dh + dq;
      uint4 u0 = *((const uint4*)gk);
      *((uint4*)(Ks + key * 64 + dq)) = u0;
      const bf16_t* gv = V + ((long)(b * S) + kkey + key) * HD + h * Dh + dq;
#pragma unroll
      for (int i = 0; i < 8; ++i) Vs[(dq + i) * 32 + key] = gv[i];
    }
    __syncthreads();

    if (kkey <= q0 + 15) {  // this wave's causal extent reaches this chunk
      // ---- S = Q K^T : preload 4 K fragments, fence, 4 WMMAs back-to-back
      v16bf kf00 = frag_ld(Ks + 0 * 64 + 0, 64);
      v16bf kf01 = frag_ld(Ks + 0 * 64 + 32, 64);
      v16bf kf10 = frag_ld(Ks + 16 * 64 + 0, 64);
      v16bf kf11 = frag_ld(Ks + 16 * 64 + 32, 64);
      SCHED_FENCE();
      v8f s0 = {}, s1 = {};
      s0 = WMMA_BF16(qf0, kf00, s0);
      s1 = WMMA_BF16(qf0, kf10, s1);
      s0 = WMMA_BF16(qf1, kf01, s0);
      s1 = WMMA_BF16(qf1, kf11, s1);
      SCHED_FENCE();

      const int n = lane & 15, mo = (lane >> 4) << 3;
#pragma unroll
      for (int r = 0; r < 8; ++r) {
        int qrow = q0 + mo + r;
        if (kkey + n > qrow)      s0[r] = -1e9f;
        if (kkey + 16 + n > qrow) s1[r] = -1e9f;
        // row max across the 16-lane group holding this score row
        float mx = fmaxf(s0[r], s1[r]);
#pragma unroll
        for (int off = 1; off < 16; off <<= 1)
          mx = fmaxf(mx, __shfl_xor(mx, off, 32));
        float mnew  = fmaxf(mrow[r], mx);
        float alpha = __expf(mrow[r] - mnew);
        float p0 = __expf(s0[r] - mnew);
        float p1 = __expf(s1[r] - mnew);
        float rs = p0 + p1;
#pragma unroll
        for (int off = 1; off < 16; off <<= 1)
          rs += __shfl_xor(rs, off, 32);
        lrow[r] = lrow[r] * alpha + rs;
        mrow[r] = mnew;
#pragma unroll
        for (int t = 0; t < 4; ++t) o[t][r] *= alpha;
        // C-layout -> LDS (A-layout round trip for the P*V matmul)
        Ps[w][(mo + r) * 32 + n]      = (bf16_t)p0;
        Ps[w][(mo + r) * 32 + 16 + n] = (bf16_t)p1;
      }
      // ---- O += P V : preload P + 4 V fragments, fence, 4 WMMAs back-to-back
      v16bf pf = frag_ld(Ps[w], 32);
      v16bf vf[4];
#pragma unroll
      for (int t = 0; t < 4; ++t) vf[t] = frag_ld(Vs + (t * 16) * 32, 32);
      SCHED_FENCE();
#pragma unroll
      for (int t = 0; t < 4; ++t) o[t] = WMMA_BF16(pf, vf[t], o[t]);
      SCHED_FENCE();
    }
    __syncthreads();
  }

  // normalize by the softmax denominator and store bf16 attn output
  {
    const int n = lane & 15, mo = (lane >> 4) << 3;
#pragma unroll
    for (int r = 0; r < 8; ++r) {
      float invl = 1.0f / lrow[r];
#pragma unroll
      for (int t = 0; t < 4; ++t)
        Oa[((long)(b * S) + q0 + mo + r) * HD + h * Dh + t * 16 + n] =
            (bf16_t)(o[t][r] * invl);
    }
  }
}

// ---------------------------------------------------------------------------
// Kernel 4: output projection (bf16 attn @ Wo -> fp32).  Same pipelined tiling.
// ---------------------------------------------------------------------------
__global__ __launch_bounds__(256)
void gemm_out_kernel(const bf16_t* __restrict__ Xa, const float* __restrict__ Wo,
                     float* __restrict__ Out) {
  const int N = 512, K = 512;
  __shared__ bf16_t As[128 * 64];  // 16 KB
  __shared__ bf16_t Bs[64 * 64];   //  8 KB
  const int bm = blockIdx.x * 128, bn = blockIdx.y * 64;
  const int tid = threadIdx.x, w = tid >> 5;

  const int arow = tid >> 1, akh = (tid & 1) * 32;
  const int bnn = tid >> 2,  bkq = (tid & 3) * 16;

  uint4 u[4];       // staged A chunk (32 bf16)
  float wvf[16];    // staged B chunk

  {
    const bf16_t* g = Xa + (long)(bm + arow) * K + akh;
#pragma unroll
    for (int i = 0; i < 4; ++i) u[i] = ((const uint4*)g)[i];
#pragma unroll
    for (int i = 0; i < 16; ++i) wvf[i] = Wo[(long)(bkq + i) * N + bn + bnn];
  }

  v8f acc[4] = {};
  for (int k0 = 0; k0 < K; k0 += 64) {
    {
      bf16_t* s = As + arow * 64 + akh;
#pragma unroll
      for (int i = 0; i < 4; ++i) *((uint4*)(s + 8 * i)) = u[i];
      bf16_t* sb = Bs + bnn * 64 + bkq;
#pragma unroll
      for (int i = 0; i < 16; ++i) sb[i] = (bf16_t)wvf[i];
    }
    __syncthreads();

    if (k0 + 64 < K) {
      const bf16_t* g = Xa + (long)(bm + arow) * K + k0 + 64 + akh;
      __builtin_prefetch(g + 64, 0, 1);
#pragma unroll
      for (int i = 0; i < 4; ++i) u[i] = ((const uint4*)g)[i];
#pragma unroll
      for (int i = 0; i < 16; ++i)
        wvf[i] = Wo[(long)(k0 + 64 + bkq + i) * N + bn + bnn];
    }

    v16bf af0 = frag_ld(As + (w * 16) * 64, 64);
    v16bf af1 = frag_ld(As + (w * 16) * 64 + 32, 64);
    v16bf bfr[8];
#pragma unroll
    for (int t = 0; t < 4; ++t) {
      bfr[2 * t]     = frag_ld(Bs + (t * 16) * 64, 64);
      bfr[2 * t + 1] = frag_ld(Bs + (t * 16) * 64 + 32, 64);
    }
    SCHED_FENCE();
#pragma unroll
    for (int t = 0; t < 4; ++t) acc[t] = WMMA_BF16(af0, bfr[2 * t], acc[t]);
#pragma unroll
    for (int t = 0; t < 4; ++t) acc[t] = WMMA_BF16(af1, bfr[2 * t + 1], acc[t]);
    SCHED_FENCE();
    __syncthreads();
  }

  const int lane = tid & 31, n = lane & 15, mo = (lane >> 4) << 3;
#pragma unroll
  for (int t = 0; t < 4; ++t)
#pragma unroll
    for (int r = 0; r < 8; ++r)
      Out[(long)(bm + w * 16 + mo + r) * N + bn + t * 16 + n] = acc[t][r];
}

// ---------------------------------------------------------------------------
extern "C" void kernel_launch(void* const* d_in, const int* in_sizes, int n_in,
                              void* d_out, int out_size, void* d_ws, size_t ws_size,
                              hipStream_t stream) {
  const float* query  = (const float*)d_in[0];
  const float* value  = (const float*)d_in[1];
  const float* key_in = (const float*)d_in[2];
  const float* Wq     = (const float*)d_in[3];
  const float* Wk     = (const float*)d_in[4];
  const float* Wv     = (const float*)d_in[5];
  const float* Wo     = (const float*)d_in[6];
  float* out = (float*)d_out;

  char* ws = (char*)d_ws;
  const size_t TS = (size_t)8192 * 512 * sizeof(bf16_t);  // 8 MB per tensor
  bf16_t* Qw = (bf16_t*)(ws);
  bf16_t* Kw = (bf16_t*)(ws + TS);
  bf16_t* Vw = (bf16_t*)(ws + 2 * TS);
  bf16_t* Aw = (bf16_t*)(ws + 3 * TS);

  // 1) QKV projections (z: 0=Q, 1=K, 2=V)
  dim3 g1(8192 / 128, 512 / 64, 3);
  gemm_qkv_kernel<<<g1, 256, 0, stream>>>(query, key_in, value, Wq, Wk, Wv,
                                          Qw, Kw, Vw);
  // 2) RoPE + q scaling
  int total = 8 * 1024 * 8 * 32;
  rope_kernel<<<(total + 255) / 256, 256, 0, stream>>>(Qw, Kw);
  // 3) causal flash attention
  dim3 g3(1024 / 128, 8 * 8);
  flash_attn_kernel<<<g3, 256, 0, stream>>>(Qw, Kw, Vw, Aw);
  // 4) output projection
  dim3 g4(8192 / 128, 512 / 64);
  gemm_out_kernel<<<g4, 256, 0, stream>>>(Aw, Wo, out);
}
